// MultiheadAttentionZSummarized_79637283603221
// MI455X (gfx1250) — matrix operand order repaired and checked by reading, not compiled, not run on hardware
//
#include <hip/hip_runtime.h>

// ---------------------------------------------------------------------------
// MultiheadAttention (q from q1*0.125, k, v = k@W^T + b), flash-attention
// style, bf16 WMMA on gfx1250 (CDNA5, wave32).
//
// Workspace layout (needs 96MB):
//   qb : bf16 [B][T][E]   (q1 * 0.125, transposed to batch-major)
//   kb : bf16 [B][S][E]
//   vb : bf16 [B][S][E]   (k @ W^T + b)
// ---------------------------------------------------------------------------

#define T_DIM 2048
#define S_DIM 2048
#define B_DIM 16
#define E_DIM 512
#define SCALING 0.125f

#define BQ 64          // query rows per block
#define BK 32          // key/value rows per S-tile
#define VT_STRIDE 40   // padded LDS stride (elements): 80B = 16B aligned, bank-conflict free
#define PT_STRIDE 40

typedef __attribute__((ext_vector_type(16))) __bf16 bf16x16;
typedef __attribute__((ext_vector_type(8)))  float  f32x8;

// A-matrix (16x32 bf16) per-lane load: 16 contiguous elements at p
__device__ __forceinline__ bf16x16 load_a16(const __bf16* p) {
  union { uint4 u[2]; bf16x16 v; } t;
  t.u[0] = *(const uint4*)(p);
  t.u[1] = *(const uint4*)(p + 8);
  return t.v;
}

// B-matrix (32x16 bf16) per-lane load: 16 contiguous elements at p
__device__ __forceinline__ bf16x16 load_b16(const __bf16* p) {
  union { uint4 u[2]; bf16x16 v; } t;
  t.u[0] = *(const uint4*)(p);
  t.u[1] = *(const uint4*)(p + 8);
  return t.v;
}

// B-matrix tile from f32 source (16 contiguous floats), converted to bf16
__device__ __forceinline__ bf16x16 cvt_b16_from_f32(const float* p) {
  float4 f0 = *(const float4*)(p + 0);
  float4 f1 = *(const float4*)(p + 4);
  float4 f2 = *(const float4*)(p + 8);
  float4 f3 = *(const float4*)(p + 12);
  bf16x16 r;
  r[0]  = (__bf16)f0.x; r[1]  = (__bf16)f0.y; r[2]  = (__bf16)f0.z; r[3]  = (__bf16)f0.w;
  r[4]  = (__bf16)f1.x; r[5]  = (__bf16)f1.y; r[6]  = (__bf16)f1.z; r[7]  = (__bf16)f1.w;
  r[8]  = (__bf16)f2.x; r[9]  = (__bf16)f2.y; r[10] = (__bf16)f2.z; r[11] = (__bf16)f2.w;
  r[12] = (__bf16)f3.x; r[13] = (__bf16)f3.y; r[14] = (__bf16)f3.z; r[15] = (__bf16)f3.w;
  return r;
}

__device__ __forceinline__ f32x8 wmma_bf16(bf16x16 a, bf16x16 b, f32x8 c) {
  return __builtin_amdgcn_wmma_f32_16x16x32_bf16(
      /*neg_a=*/false, a, /*neg_b=*/false, b,
      /*c_mod=*/(short)0, c, /*reuse_a=*/false, /*reuse_b=*/false);
}

// ---------------------------------------------------------------------------
// Kernel 1: f32 -> bf16 staging.  [T,B,E] -> [B,T,E], q scaled.
// ---------------------------------------------------------------------------
__global__ __launch_bounds__(256) void prep_kernel(
    const float* __restrict__ q1, const float* __restrict__ k,
    __bf16* __restrict__ qb, __bf16* __restrict__ kb) {
  const long long gid  = (long long)blockIdx.x * blockDim.x + threadIdx.x;
  const long long base = gid * 8;                      // flat [T,B,E] index
  const int e = (int)(base & (E_DIM - 1));
  const long long rb = base >> 9;                      // t*B + b
  const int b = (int)(rb & (B_DIM - 1));
  const int t = (int)(rb >> 4);

  float4 x0 = *(const float4*)(q1 + base);
  float4 x1 = *(const float4*)(q1 + base + 4);
  float4 y0 = *(const float4*)(k  + base);
  float4 y1 = *(const float4*)(k  + base + 4);

  union { uint4 u; __bf16 h[8]; } qo, ko;
  qo.h[0] = (__bf16)(x0.x * SCALING); qo.h[1] = (__bf16)(x0.y * SCALING);
  qo.h[2] = (__bf16)(x0.z * SCALING); qo.h[3] = (__bf16)(x0.w * SCALING);
  qo.h[4] = (__bf16)(x1.x * SCALING); qo.h[5] = (__bf16)(x1.y * SCALING);
  qo.h[6] = (__bf16)(x1.z * SCALING); qo.h[7] = (__bf16)(x1.w * SCALING);
  ko.h[0] = (__bf16)y0.x; ko.h[1] = (__bf16)y0.y;
  ko.h[2] = (__bf16)y0.z; ko.h[3] = (__bf16)y0.w;
  ko.h[4] = (__bf16)y1.x; ko.h[5] = (__bf16)y1.y;
  ko.h[6] = (__bf16)y1.z; ko.h[7] = (__bf16)y1.w;

  const long long dst = ((long long)b * T_DIM + t) * E_DIM + e;
  *(uint4*)(qb + dst) = qo.u;
  *(uint4*)(kb + dst) = ko.u;
}

// ---------------------------------------------------------------------------
// Kernel 2: v = k @ W^T + bias, bf16 WMMA.  Rows r = b*S + s (flattened),
// block computes 64 rows x 64 cols; 8 waves = 4 row-groups x 2 col-groups,
// each wave owns two 16x16 C tiles.
// ---------------------------------------------------------------------------
__global__ __launch_bounds__(256) void vproj_kernel(
    const __bf16* __restrict__ kb, const float* __restrict__ W,
    const float* __restrict__ bias, __bf16* __restrict__ vb) {
  const int tid  = threadIdx.x;
  const int lane = tid & 31, warp = tid >> 5;
  const int wm = warp & 3, wn = warp >> 2;
  const int hi = lane >> 4;
  const int ln = lane & 15;

  const int r0 = blockIdx.y * 64 + wm * 16;
  const int e0 = blockIdx.x * 64 + wn * 32;

  const __bf16* aptr = kb + (long long)(r0 + ln) * E_DIM + hi * 16;
  const float*  w0   = W + (long long)(e0 + ln) * E_DIM + hi * 16;
  const float*  w1   = W + (long long)(e0 + 16 + ln) * E_DIM + hi * 16;

  f32x8 c0 = {0.f,0.f,0.f,0.f,0.f,0.f,0.f,0.f};
  f32x8 c1 = {0.f,0.f,0.f,0.f,0.f,0.f,0.f,0.f};

  #pragma unroll
  for (int d0 = 0; d0 < E_DIM; d0 += 32) {
    bf16x16 a  = load_a16(aptr + d0);
    bf16x16 b0 = cvt_b16_from_f32(w0 + d0);
    c0 = wmma_bf16(a, b0, c0);
    bf16x16 b1 = cvt_b16_from_f32(w1 + d0);
    c1 = wmma_bf16(a, b1, c1);
  }

  const float bv0 = bias[e0 + ln];
  const float bv1 = bias[e0 + 16 + ln];
  #pragma unroll
  for (int j = 0; j < 8; ++j) {
    const long long row = r0 + j + hi * 8;
    vb[row * E_DIM + e0 + ln]      = (__bf16)(c0[j] + bv0);
    vb[row * E_DIM + e0 + 16 + ln] = (__bf16)(c1[j] + bv1);
  }
}

// ---------------------------------------------------------------------------
// Kernel 3: flash attention.  Block = (64 q rows, one batch), 8 waves:
// wm in 0..3 -> 16-row group, wn in 0..1 -> 256-col half of E for output.
// ---------------------------------------------------------------------------
__global__ __launch_bounds__(256) void flash_kernel(
    const __bf16* __restrict__ qb, const __bf16* __restrict__ kb,
    const __bf16* __restrict__ vb, const unsigned char* __restrict__ mask,
    float* __restrict__ out) {
  __shared__ __align__(16) __bf16 VT[E_DIM * VT_STRIDE];   // V^T tile [e][s], padded
  __shared__ __align__(16) __bf16 PT[8][16 * PT_STRIDE];   // per-wave P tile [row][col]

  const int tid  = threadIdx.x;
  const int lane = tid & 31, warp = tid >> 5;
  const int wm = warp & 3, wn = warp >> 2;
  const int hi = lane >> 4;
  const int ln = lane & 15;

  const int b  = blockIdx.y;
  const int t0 = blockIdx.x * BQ;
  const long long brow = (long long)b * S_DIM;

  // Cooperative V-stage mapping: each thread owns an 8x8 bf16 block:
  // 8 s-rows (sgrp) x 8 e-cols (e8), transposed in registers.
  const int sgrp = (tid & 3) * 8;          // 0,8,16,24
  const int e8   = (tid >> 2) * 8;         // 0..504

  // Q tile resident in registers: 16 rows x 512 K, A-layout per 32-K chunk.
  bf16x16 qa[16];
  {
    const __bf16* qp = qb + ((long long)b * T_DIM + (t0 + wm * 16 + ln)) * E_DIM + hi * 16;
    #pragma unroll
    for (int c = 0; c < 16; ++c) qa[c] = load_a16(qp + c * 32);
  }

  f32x8 oacc[16];
  #pragma unroll
  for (int nt = 0; nt < 16; ++nt)
    oacc[nt] = (f32x8){0.f,0.f,0.f,0.f,0.f,0.f,0.f,0.f};
  float mstate[8], lstate[8];
  #pragma unroll
  for (int j = 0; j < 8; ++j) { mstate[j] = -1e30f; lstate[j] = 0.f; }

  for (int s0 = 0; s0 < S_DIM; s0 += BK) {
    // Prefetch next S-tile's K and V rows (global_prefetch_b8 path).
    const int s1 = s0 + BK;
    if (s1 < S_DIM) {
      const char* nk = (const char*)(kb + (brow + s1) * E_DIM);
      const char* nv = (const char*)(vb + (brow + s1) * E_DIM);
      __builtin_prefetch(nk + tid * 128, 0, 3);   // 256 threads x 128B = 32KB tile
      __builtin_prefetch(nv + tid * 128, 0, 3);
    }

    __syncthreads();   // previous iteration's VT reads complete

    // Cooperative load of V tile [BK x E]: 8x8 register transpose, b128 LDS stores.
    {
      union RowU { uint4 u; __bf16 h[8]; };
      RowU in[8];
      #pragma unroll
      for (int r = 0; r < 8; ++r)
        in[r].u = *(const uint4*)(vb + (brow + s0 + sgrp + r) * E_DIM + e8);
      #pragma unroll
      for (int j = 0; j < 8; ++j) {
        RowU o;
        #pragma unroll
        for (int r = 0; r < 8; ++r) o.h[r] = in[r].h[j];
        *(uint4*)&VT[(e8 + j) * VT_STRIDE + sgrp] = o.u;
      }
    }

    // Scores: 16x32 per wave (two 16x16 C tiles), K tiles straight from global.
    f32x8 sc0 = {0.f,0.f,0.f,0.f,0.f,0.f,0.f,0.f};
    f32x8 sc1 = {0.f,0.f,0.f,0.f,0.f,0.f,0.f,0.f};
    const __bf16* kp0 = kb + (brow + s0 + ln) * E_DIM + hi * 16;
    const __bf16* kp1 = kb + (brow + s0 + 16 + ln) * E_DIM + hi * 16;
    #pragma unroll
    for (int c = 0; c < 16; ++c) {
      bf16x16 b0 = load_b16(kp0 + c * 32);
      sc0 = wmma_bf16(qa[c], b0, sc0);
      bf16x16 b1 = load_b16(kp1 + c * 32);
      sc1 = wmma_bf16(qa[c], b1, sc1);
    }

    __syncthreads();   // VT writes visible to all waves

    // Key-padding mask: column s masked -> score -inf, prob 0.
    const unsigned char mk0 = mask[b * S_DIM + s0 + ln];
    const unsigned char mk1 = mask[b * S_DIM + s0 + 16 + ln];
    if (mk0) {
      #pragma unroll
      for (int j = 0; j < 8; ++j) sc0[j] = -1e30f;
    }
    if (mk1) {
      #pragma unroll
      for (int j = 0; j < 8; ++j) sc1[j] = -1e30f;
    }

    // Online softmax per row (row = j + hi*8; 16-lane groups share a row).
    float scl[8];
    #pragma unroll
    for (int j = 0; j < 8; ++j) {
      float mx = fmaxf(sc0[j], sc1[j]);
      mx = fmaxf(mx, __shfl_xor(mx, 1, 32));
      mx = fmaxf(mx, __shfl_xor(mx, 2, 32));
      mx = fmaxf(mx, __shfl_xor(mx, 4, 32));
      mx = fmaxf(mx, __shfl_xor(mx, 8, 32));
      const float mnew = fmaxf(mstate[j], mx);
      const float s    = __expf(mstate[j] - mnew);
      const float p0   = mk0 ? 0.f : __expf(sc0[j] - mnew);
      const float p1   = mk1 ? 0.f : __expf(sc1[j] - mnew);
      float rs = p0 + p1;
      rs += __shfl_xor(rs, 1, 32);
      rs += __shfl_xor(rs, 2, 32);
      rs += __shfl_xor(rs, 4, 32);
      rs += __shfl_xor(rs, 8, 32);
      lstate[j] = lstate[j] * s + rs;
      mstate[j] = mnew;
      scl[j] = s;
      const int prow = j + hi * 8;
      PT[warp][prow * PT_STRIDE + ln]      = (__bf16)p0;
      PT[warp][prow * PT_STRIDE + 16 + ln] = (__bf16)p1;
    }

    // Rescale running output.
    #pragma unroll
    for (int nt = 0; nt < 16; ++nt) {
      #pragma unroll
      for (int j = 0; j < 8; ++j) oacc[nt][j] *= scl[j];
    }

    // P in A-layout from per-wave LDS (same-wave DS ops are in-order).
    const bf16x16 pa = load_a16(&PT[warp][ln * PT_STRIDE + hi * 16]);

    // O += P @ V ; V B-tiles from transposed LDS (contiguous, conflict-free).
    #pragma unroll
    for (int nt = 0; nt < 16; ++nt) {
      const __bf16* vp = &VT[(wn * 256 + nt * 16 + ln) * VT_STRIDE + hi * 16];
      bf16x16 vt = load_b16(vp);
      oacc[nt] = wmma_bf16(pa, vt, oacc[nt]);
    }
  }

  // Epilogue: out[t, b, e] = O / l
  #pragma unroll
  for (int nt = 0; nt < 16; ++nt) {
    const int col = wn * 256 + nt * 16 + ln;
    #pragma unroll
    for (int j = 0; j < 8; ++j) {
      const int trow = t0 + wm * 16 + j + hi * 8;
      out[((long long)trow * B_DIM + b) * E_DIM + col] = oacc[nt][j] / lstate[j];
    }
  }
}

// ---------------------------------------------------------------------------
extern "C" void kernel_launch(void* const* d_in, const int* in_sizes, int n_in,
                              void* d_out, int out_size, void* d_ws, size_t ws_size,
                              hipStream_t stream) {
  (void)in_sizes; (void)n_in; (void)out_size; (void)ws_size;

  const float* q1 = (const float*)d_in[0];
  const float* k  = (const float*)d_in[1];
  // d_in[2] = value (unused by reference)
  const unsigned char* mask = (const unsigned char*)d_in[3];
  const float* W    = (const float*)d_in[4];
  const float* bias = (const float*)d_in[5];
  float* out = (float*)d_out;

  const size_t planeBytes = (size_t)B_DIM * T_DIM * E_DIM * sizeof(__bf16); // 32MB
  char* ws = (char*)d_ws;
  __bf16* qb = (__bf16*)(ws);
  __bf16* kb = (__bf16*)(ws + planeBytes);
  __bf16* vb = (__bf16*)(ws + 2 * planeBytes);

  const int total   = B_DIM * T_DIM * E_DIM;       // 16,777,216
  const int pblocks = total / (256 * 8);           // 8192
  prep_kernel<<<pblocks, 256, 0, stream>>>(q1, k, qb, kb);

  vproj_kernel<<<dim3(E_DIM / 64, (B_DIM * S_DIM) / 64), 256, 0, stream>>>(kb, W, bias, vb);

  flash_kernel<<<dim3(T_DIM / BQ, B_DIM), 256, 0, stream>>>(qb, kb, vb, mask, out);
}
